// SelfAttention_71786083386189
// MI455X (gfx1250) — matrix operand order repaired and checked by reading, not compiled, not run on hardware
//
#include <hip/hip_runtime.h>
#include <hip/hip_bf16.h>

// ---------------------------------------------------------------------------
// Causal self-attention, single head.  B=4, T=4096, D=1024, HD=64 (f32 io).
// Kernel 1: QKV projection via v_wmma_f32_16x16x32_bf16 (x,W converted to bf16
//           in-register).  Q scaled by 1/sqrt(HD).  V stored transposed.
// Kernel 2: split-KV flash attention: 4 waves share one 16-row query tile,
//           each wave owns kv-chunks strided by 128; partial (O,m,l) merged
//           in LDS.  All matmuls (S = Q K^T, rowsum(P), O += P V) use bf16
//           WMMA with f32 accumulation.  global_prefetch for next kv chunk.
// Workspace: Qbf (2MB) | Kbf (2MB) | Vt bf (2MB) = 6 MB  (fits in L2).
// ---------------------------------------------------------------------------

typedef __attribute__((ext_vector_type(16))) __bf16 v16bf;
typedef __attribute__((ext_vector_type(8)))  float  v8f;
typedef __attribute__((ext_vector_type(4)))  int    i32x4;
typedef __attribute__((ext_vector_type(4)))  float  f32x4;

static constexpr int Bc = 4, Tc = 4096, Dc = 1024, HDc = 64;

__device__ __forceinline__ v8f wmma_bf16(v16bf a, v16bf b, v8f c) {
  return __builtin_amdgcn_wmma_f32_16x16x32_bf16(false, a, false, b,
                                                 (short)0, c, false, false);
}

// Load 16 bf16 as two 16B chunks (A and B frag: p0=K run0, p1=K run1 = p0+16)
__device__ __forceinline__ v16bf load16bf(const __bf16* p0, const __bf16* p1) {
  union { v16bf v; i32x4 q[2]; } u;
  u.q[0] = *(const i32x4*)p0;
  u.q[1] = *(const i32x4*)p1;
  return u.v;
}

// Load 2x8 contiguous f32 and convert to a 16-element bf16 fragment.
__device__ __forceinline__ v16bf cvt8x2(const float* p0, const float* p1) {
  f32x4 a0 = ((const f32x4*)p0)[0];
  f32x4 a1 = ((const f32x4*)p0)[1];
  f32x4 b0 = ((const f32x4*)p1)[0];
  f32x4 b1 = ((const f32x4*)p1)[1];
  v16bf v;
#pragma unroll
  for (int i = 0; i < 4; ++i) {
    v[i]      = (__bf16)a0[i];
    v[4 + i]  = (__bf16)a1[i];
    v[8 + i]  = (__bf16)b0[i];
    v[12 + i] = (__bf16)b1[i];
  }
  return v;
}

// ---------------------------------------------------------------------------
// Kernel 1: QKV projection.  One wave computes one 16x16 output tile.
// gw = rt*12 + (mat*4 + nt);  rt over B*T/16 row tiles.
// ---------------------------------------------------------------------------
__global__ void __launch_bounds__(128)
qkv_proj_kernel(const float* __restrict__ x,
                const float* __restrict__ Wq,
                const float* __restrict__ Wk,
                const float* __restrict__ Wv,
                __bf16* __restrict__ qb,   // [B*T, 64]
                __bf16* __restrict__ kb,   // [B*T, 64]
                __bf16* __restrict__ vtb)  // [B, 64, T]
{
  const int lane = threadIdx.x & 31;
  const int wave = threadIdx.x >> 5;
  const int gw   = blockIdx.x * 4 + wave;
  const int rt   = gw / 12;
  const int c    = gw % 12;
  const int mat  = c >> 2;   // 0=Q 1=K 2=V
  const int nt   = c & 3;    // head-dim 16-col tile
  const int n    = lane & 15;
  const int half = lane >> 4;

  const float* W = (mat == 0) ? Wq : (mat == 1) ? Wk : Wv;
  const float* wrow = W + (size_t)(nt * 16 + n) * Dc;        // B-frag row
  const float* xrow = x + (size_t)(rt * 16 + n) * Dc;        // A-frag row (m=n)

  v8f acc;
#pragma unroll
  for (int r = 0; r < 8; ++r) acc[r] = 0.0f;

  for (int ks = 0; ks < Dc / 32; ++ks) {
    const int k0 = ks * 32;
    v16bf a = cvt8x2(xrow + k0 + half * 8, xrow + k0 + 16 + half * 8);
    v16bf b = cvt8x2(wrow + k0 + half * 8, wrow + k0 + 16 + half * 8);
    acc = wmma_bf16(a, b, acc);
  }

  if (mat == 0) {
    const float s = 0.125f;  // 1/sqrt(64) folded into Q
#pragma unroll
    for (int r = 0; r < 8; ++r)
      qb[(size_t)(rt * 16 + 8 * half + r) * HDc + nt * 16 + n] =
          (__bf16)(acc[r] * s);
  } else if (mat == 1) {
#pragma unroll
    for (int r = 0; r < 8; ++r)
      kb[(size_t)(rt * 16 + 8 * half + r) * HDc + nt * 16 + n] = (__bf16)acc[r];
  } else {
    const int b    = rt >> 8;            // 256 row-tiles per batch
    const int trow = (rt & 255) * 16;
    __bf16* vb = vtb + (size_t)b * HDc * Tc;
#pragma unroll
    for (int r = 0; r < 8; ++r)
      vb[(size_t)(nt * 16 + n) * Tc + trow + 8 * half + r] = (__bf16)acc[r];
  }
}

// ---------------------------------------------------------------------------
// Kernel 2: split-KV flash attention.  One 4-wave block per 16 query rows;
// wave w handles kv chunks  w*32, w*32+128, ...  (strided -> causal balance).
// ---------------------------------------------------------------------------
__global__ void __launch_bounds__(128)
flash_attn_kernel(const __bf16* __restrict__ qb,
                  const __bf16* __restrict__ kb,
                  const __bf16* __restrict__ vtb,
                  float* __restrict__ out)
{
  // per-wave 16x32 bf16 P tile, row stride padded to 40 halves (80B)
  __shared__ __bf16 plds[4 * 16 * 40];
  // split-KV merge buffers
  __shared__ float lacc[4][16][64];   // partial O per wave
  __shared__ float lm[4][16], ll[4][16];
  __shared__ float lws[4][16], linv[16];

  const int lane   = threadIdx.x & 31;
  const int wave   = threadIdx.x >> 5;
  const int qt     = blockIdx.x;
  const int b      = qt >> 8;
  const int t_base = (qt & 255) * 16;
  const int n      = lane & 15;
  const int half   = lane >> 4;
  __bf16* pl = plds + wave * 16 * 40;

  // Resident Q A-fragments (16x64 = two 16x32 frags), scale pre-folded.
  const __bf16* qrow = qb + (size_t)(b * Tc + t_base + n) * HDc;
  const v16bf a0 = load16bf(qrow + half * 8,      qrow + 16 + half * 8);
  const v16bf a1 = load16bf(qrow + 32 + half * 8, qrow + 48 + half * 8);

  v16bf ones;
#pragma unroll
  for (int i = 0; i < 16; ++i) ones[i] = (__bf16)1.0f;

  float mrow[8], lrow[8];
  v8f acc[4];
#pragma unroll
  for (int r = 0; r < 8; ++r) { mrow[r] = -1e30f; lrow[r] = 0.0f; }
#pragma unroll
  for (int t = 0; t < 4; ++t)
#pragma unroll
    for (int r = 0; r < 8; ++r) acc[t][r] = 0.0f;

  const __bf16* kbase = kb + (size_t)b * Tc * HDc;
  const __bf16* vbase = vtb + (size_t)b * HDc * Tc;

  for (int kv = wave * 32; kv < t_base + 16; kv += 128) {
    // ---- S = Q K^T : two 16x16 tiles over K-dim 64 --------------------
    const __bf16* kr0 = kbase + (size_t)(kv + n) * HDc;
    const __bf16* kr1 = kbase + (size_t)(kv + 16 + n) * HDc;
    // prefetch this wave's next chunk (kv+128) K rows into cache
    __builtin_prefetch(kr0 + (size_t)128 * HDc, 0, 1);
    __builtin_prefetch(kr1 + (size_t)128 * HDc, 0, 1);

    v8f s0, s1;
#pragma unroll
    for (int r = 0; r < 8; ++r) { s0[r] = 0.0f; s1[r] = 0.0f; }
    s0 = wmma_bf16(a0, load16bf(kr0 + half * 8,      kr0 + 16 + half * 8),  s0);
    s0 = wmma_bf16(a1, load16bf(kr0 + 32 + half * 8, kr0 + 48 + half * 8), s0);
    s1 = wmma_bf16(a0, load16bf(kr1 + half * 8,      kr1 + 16 + half * 8),  s1);
    s1 = wmma_bf16(a1, load16bf(kr1 + 32 + half * 8, kr1 + 48 + half * 8), s1);

    // ---- causal mask on diagonal chunk (wave-uniform branch) ----------
    if (kv + 31 > t_base) {
#pragma unroll
      for (int r = 0; r < 8; ++r) {
        const int qg = t_base + 8 * half + r;
        if (kv + n > qg)      s0[r] = -1e30f;
        if (kv + 16 + n > qg) s1[r] = -1e30f;
      }
    }

    // ---- online softmax: row max across the 16 lanes of each half -----
    float scale[8];
#pragma unroll
    for (int r = 0; r < 8; ++r) {
      float mx = fmaxf(s0[r], s1[r]);
      mx = fmaxf(mx, __shfl_xor(mx, 1, 16));
      mx = fmaxf(mx, __shfl_xor(mx, 2, 16));
      mx = fmaxf(mx, __shfl_xor(mx, 4, 16));
      mx = fmaxf(mx, __shfl_xor(mx, 8, 16));
      const float mnew = fmaxf(mrow[r], mx);
      scale[r] = __expf(mrow[r] - mnew);
      mrow[r]  = mnew;
      // exp -> bf16, D-layout -> LDS (row-major 16x32, padded stride 40)
      pl[(8 * half + r) * 40 + n]      = (__bf16)__expf(s0[r] - mnew);
      pl[(8 * half + r) * 40 + 16 + n] = (__bf16)__expf(s1[r] - mnew);
    }
    asm volatile("s_wait_dscnt 0" ::: "memory");  // wave-private LDS bounce

    // ---- P as A-fragment (row m = lane%16, K runs at half*8, +16) -----
    const __bf16* pr = pl + n * 40;
    const v16bf pa = load16bf(pr + half * 8, pr + 16 + half * 8);

    // ---- row sums via WMMA against all-ones B -------------------------
    v8f ls;
#pragma unroll
    for (int r = 0; r < 8; ++r) ls[r] = 0.0f;
    ls = wmma_bf16(pa, ones, ls);
#pragma unroll
    for (int r = 0; r < 8; ++r) lrow[r] = lrow[r] * scale[r] + ls[r];

    // ---- rescale accumulators, then O += P V --------------------------
#pragma unroll
    for (int t = 0; t < 4; ++t)
#pragma unroll
      for (int r = 0; r < 8; ++r) acc[t][r] *= scale[r];

#pragma unroll
    for (int t = 0; t < 4; ++t) {
      const __bf16* vr = vbase + (size_t)(t * 16 + n) * Tc + kv;
      __builtin_prefetch(vr + 128, 0, 1);  // next chunk along T
      acc[t] = wmma_bf16(pa, load16bf(vr + half * 8, vr + 16 + half * 8),
                         acc[t]);
    }
  }

  // ---- write partial (O, m, l) to LDS ---------------------------------
#pragma unroll
  for (int t = 0; t < 4; ++t)
#pragma unroll
    for (int r = 0; r < 8; ++r)
      lacc[wave][8 * half + r][t * 16 + n] = acc[t][r];
  if (n == 0) {
#pragma unroll
    for (int r = 0; r < 8; ++r) {
      lm[wave][8 * half + r] = mrow[r];
      ll[wave][8 * half + r] = lrow[r];
    }
  }
  __syncthreads();

  // ---- per-row merge factors ------------------------------------------
  if (threadIdx.x < 16) {
    const int row = threadIdx.x;
    float mstar = fmaxf(fmaxf(lm[0][row], lm[1][row]),
                        fmaxf(lm[2][row], lm[3][row]));
    float lsum = 0.0f;
#pragma unroll
    for (int w = 0; w < 4; ++w) {
      const float ew = __expf(lm[w][row] - mstar);
      lws[w][row] = ew;
      lsum += ll[w][row] * ew;
    }
    linv[row] = 1.0f / lsum;
  }
  __syncthreads();

  // ---- combine partials and store -------------------------------------
#pragma unroll
  for (int k = 0; k < 8; ++k) {
    const int e   = threadIdx.x + k * 128;   // 16*64 elements / 128 threads
    const int row = e >> 6;
    const int col = e & 63;
    const float v = lacc[0][row][col] * lws[0][row] +
                    lacc[1][row][col] * lws[1][row] +
                    lacc[2][row][col] * lws[2][row] +
                    lacc[3][row][col] * lws[3][row];
    out[(size_t)(b * Tc + t_base + row) * HDc + col] = v * linv[row];
  }
}

// ---------------------------------------------------------------------------
extern "C" void kernel_launch(void* const* d_in, const int* in_sizes, int n_in,
                              void* d_out, int out_size, void* d_ws,
                              size_t ws_size, hipStream_t stream) {
  const float* x  = (const float*)d_in[0];
  const float* Wq = (const float*)d_in[1];
  const float* Wk = (const float*)d_in[2];
  const float* Wv = (const float*)d_in[3];
  float* out = (float*)d_out;

  __bf16* qb  = (__bf16*)d_ws;                       // 2 MB
  __bf16* kbf = qb + (size_t)Bc * Tc * HDc;          // 2 MB
  __bf16* vtb = kbf + (size_t)Bc * Tc * HDc;         // 2 MB

  // Kernel 1: B*T/16 row tiles * 12 (mat,ntile) combos, 4 waves / block
  const int waves1 = (Bc * Tc / 16) * 12;
  qkv_proj_kernel<<<waves1 / 4, 128, 0, stream>>>(x, Wq, Wk, Wv, qb, kbf, vtb);

  // Kernel 2: one 4-wave block per 16 query rows (split-KV inside block)
  const int qtiles = Bc * Tc / 16;
  flash_attn_kernel<<<qtiles, 128, 0, stream>>>(qb, kbf, vtb, out);
}
